// ComparisonBlock_70583492542479
// MI455X (gfx1250) — compile-verified, matrix operand order checked by reading, not run on hardware
//
#include <hip/hip_runtime.h>
#include <hip/hip_bf16.h>
#include <math.h>

// ---------------------------------------------------------------------------
// Fused LN -> Q/K projection -> Q@K^T -> softmax for MI455X (gfx1250).
// Matrix work uses v_wmma_f32_16x16x32_f16 (wave32 WMMA), f32 accumulate.
// K-loops are fully unrolled (24 constant steps): no loop backedge means the
// register allocator never needs rotation copies, and the pressure-aware
// scheduler hoists global/LDS loads ahead of the WMMAs that consume them.
// ---------------------------------------------------------------------------

typedef __attribute__((ext_vector_type(16))) _Float16 v16h;
typedef __attribute__((ext_vector_type(8)))  _Float16 h8;
typedef __attribute__((ext_vector_type(8)))  float    v8f;
typedef __attribute__((ext_vector_type(4)))  int      v4i;

#define CAT16(lo, hi) __builtin_shufflevector((lo), (hi), \
    0,1,2,3,4,5,6,7,8,9,10,11,12,13,14,15)

#define WMMA_F16(A, Bm, C) __builtin_amdgcn_wmma_f32_16x16x32_f16( \
    false, (A), false, (Bm), (short)0, (C), false, false)

#if __has_builtin(__builtin_amdgcn_global_load_async_to_lds_b128) && \
    __has_builtin(__builtin_amdgcn_s_wait_asynccnt)
#define HAVE_ASYNC_LDS 1
typedef __attribute__((address_space(1))) v4i as1_v4i;   // global int4
typedef __attribute__((address_space(3))) v4i as3_v4i;   // LDS int4
#else
#define HAVE_ASYNC_LDS 0
#endif

constexpr int D  = 768;      // embedding dim (K of both GEMMs)
constexpr int S  = 2048;     // sequence length
constexpr int NB = 8;        // batch
constexpr int M  = NB * S;   // 16384 fused rows
constexpr int TSTRIDE = 16 * D;   // 12288 elements between 16-row tiles
constexpr int KSTEPS  = D / 32;   // 24 WMMA k-steps

__device__ __forceinline__ v16h ld_tile16(const _Float16* __restrict__ p) {
  // Per-lane A/B fragment: two contiguous 8xf16 chunks (16B each) at +0, +16.
  return CAT16(*(const h8*)p, *(const h8*)(p + 16));
}

// ---------------------------------------------------------------------------
// Kernel 0: weight convert + transpose.  wq is [D_in, D_out] row-major f32;
// we want wT[N=D_out][K=D_in] f16 so a B-matrix lane (one output column) reads
// contiguous 16-byte chunks of K.
// ---------------------------------------------------------------------------
__global__ __launch_bounds__(256) void wT_f16_kernel(
    const float* __restrict__ w, _Float16* __restrict__ wT) {
  int idx = blockIdx.x * 256 + threadIdx.x;
  if (idx < D * D) {
    int d = idx / D;          // input dim (K)
    int e = idx % D;          // output dim (N)
    wT[(size_t)e * D + d] = (_Float16)w[idx];
  }
}

// ---------------------------------------------------------------------------
// Kernel 1: LayerNorm over last dim, emit f16 hn[M x D].
// One 256-thread block per row; 3 elements per thread; shared-mem tree reduce.
// ---------------------------------------------------------------------------
__global__ __launch_bounds__(256) void ln_kernel(
    const float* __restrict__ h, const float* __restrict__ gamma,
    const float* __restrict__ beta, _Float16* __restrict__ hn) {
  __shared__ float s1[256];
  __shared__ float s2[256];
  const int row = blockIdx.x;
  const int t   = threadIdx.x;
  const float* x = h + (size_t)row * D;
  float a0 = x[t], a1 = x[t + 256], a2 = x[t + 512];
  s1[t] = a0 + a1 + a2;
  s2[t] = a0 * a0 + a1 * a1 + a2 * a2;
  __syncthreads();
  for (int off = 128; off > 0; off >>= 1) {
    if (t < off) { s1[t] += s1[t + off]; s2[t] += s2[t + off]; }
    __syncthreads();
  }
  const float mu  = s1[0] * (1.0f / D);
  const float var = s2[0] * (1.0f / D) - mu * mu;
  const float rs  = rsqrtf(var + 1e-5f);
  _Float16* o = hn + (size_t)row * D;
  o[t]       = (_Float16)((a0 - mu) * rs * gamma[t]       + beta[t]);
  o[t + 256] = (_Float16)((a1 - mu) * rs * gamma[t + 256] + beta[t + 256]);
  o[t + 512] = (_Float16)((a2 - mu) * rs * gamma[t + 512] + beta[t + 512]);
}

// ---------------------------------------------------------------------------
// Kernel 2: projection GEMM  out[M,N=768] = hn[M,768] @ wT^T + bias, f16 out.
// One wave owns a 16x64 strip (A reused x4).  Fully unrolled k-loop: the
// scheduler interleaves the 24 steps' loads and WMMAs without copies.
// 12288 wave-jobs = 1024 m-tiles x 12 column groups.
// ---------------------------------------------------------------------------
__global__ __launch_bounds__(256) void gemm_qk_kernel(
    const _Float16* __restrict__ A, const _Float16* __restrict__ BT,
    const float* __restrict__ bias, _Float16* __restrict__ out) {
  const int lane  = threadIdx.x & 31;
  const int wid   = blockIdx.x * 8 + (threadIdx.x >> 5);
  const int mtile = wid / 12;                 // 0..1023
  const int ng    = wid % 12;                 // 0..11 (64 cols each)
  const int r     = lane & 15;
  const int kb    = (lane < 16) ? 0 : 8;
  const int n0    = ng * 64 + r;

  const _Float16* ap = A  + (size_t)(mtile * 16 + r) * D + kb;
  const _Float16* bp = BT + (size_t)n0 * D + kb;

  v8f acc[4] = {};
#pragma unroll
  for (int s = 0; s < KSTEPS; ++s) {
    const int kk = 32 * s;
    v16h a = ld_tile16(ap + kk);
#pragma unroll
    for (int t = 0; t < 4; ++t) {
      v16h bm = ld_tile16(bp + t * TSTRIDE + kk);
      acc[t] = WMMA_F16(a, bm, acc[t]);
    }
  }

  // C/D layout: lanes 0-15 hold rows 0..7 (vgpr i -> M=i), lanes 16-31 rows 8..15.
  const int mbase = mtile * 16 + ((lane < 16) ? 0 : 8);
#pragma unroll
  for (int t = 0; t < 4; ++t) {
    const int n  = n0 + 16 * t;
    const float bv = bias[n];
#pragma unroll
    for (int i = 0; i < 8; ++i)
      out[(size_t)(mbase + i) * D + n] = (_Float16)(acc[t][i] + bv);
  }
}

// ---------------------------------------------------------------------------
// Kernel 3: fused scores + softmax.
// Block = (batch b, 16 query rows).  q-strip (16x768 f16, 24 KiB) staged in
// LDS (async-to-LDS, tracked via ASYNCcnt) and fed to the WMMA A operand via
// ds_load_b128.  B operand columns of q@k^T are k rows -> contiguous f16
// global loads.  Fully unrolled k-loop per tile group.  Score strip 16x2048
// f32 (128 KiB) lives in LDS (CDNA5: 320 KB/WGP), then block-wide
// max/exp/sum/normalize and a single streamed f32 store.
// ---------------------------------------------------------------------------
__global__ __launch_bounds__(256) void scores_softmax_kernel(
    const _Float16* __restrict__ q, const _Float16* __restrict__ k,
    float* __restrict__ out) {
  extern __shared__ __align__(16) char smem[];
  _Float16* aT   = (_Float16*)smem;                      // 24576 B
  float*    sc   = (float*)(smem + 24576);               // 131072 B
  float*    red  = (float*)(smem + 24576 + 131072);      // 1024 B
  float*    rowv = red + 256;                            // 64 B

  const int tid = threadIdx.x;
  const int b   = blockIdx.x >> 7;     // batch
  const int mt  = blockIdx.x & 127;    // 16-row strip within sequence
  const size_t qbase = (size_t)(b * S + mt * 16) * D;

  // Stage the q strip into LDS: 1536 x 16B chunks.
#if HAVE_ASYNC_LDS
  {
    const h8* gsrc = (const h8*)(q + qbase);
    h8* ldst = (h8*)aT;
    for (int ci = tid; ci < 16 * D / 8; ci += 256)
      __builtin_amdgcn_global_load_async_to_lds_b128(
          (as1_v4i*)(gsrc + ci), (as3_v4i*)(ldst + ci), 0, 0);
    __builtin_amdgcn_s_wait_asynccnt(0);
  }
#else
  for (int ci = tid; ci < 16 * D / 8; ci += 256)
    ((h8*)aT)[ci] = ((const h8*)(q + qbase))[ci];
#endif
  __syncthreads();

  const int lane = tid & 31;
  const int wave = tid >> 5;
  const int r    = lane & 15;
  const int kb   = (lane < 16) ? 0 : 8;
  const int rbase = (lane < 16) ? 0 : 8;
  const _Float16* arow = aT + r * D + kb;

  for (int g = 0; g < 4; ++g) {                    // 4 groups of 4 tiles
    const int ncol0 = wave * 256 + g * 64;
    const _Float16* kp = k + (size_t)(b * S + ncol0 + r) * D + kb;

    v8f acc[4] = {};
#pragma unroll
    for (int s = 0; s < KSTEPS; ++s) {
      const int kk = 32 * s;
      v16h a = ld_tile16(arow + kk);
#pragma unroll
      for (int t = 0; t < 4; ++t) {
        v16h bm = ld_tile16(kp + t * TSTRIDE + kk);
        acc[t] = WMMA_F16(a, bm, acc[t]);
      }
    }
#pragma unroll
    for (int t = 0; t < 4; ++t)
#pragma unroll
      for (int i = 0; i < 8; ++i)
        sc[(rbase + i) * 2048 + ncol0 + 16 * t + r] = acc[t][i];
  }
  __syncthreads();

  // Softmax over each of the 16 rows of sc[16][2048]; 16 threads per row.
  const int rr  = tid >> 4;      // row 0..15
  const int seg = tid & 15;      // 128-column segment
  float* srow = sc + rr * 2048 + seg * 128;

  float mx = -INFINITY;
  for (int c = 0; c < 128; ++c) mx = fmaxf(mx, srow[c]);
  red[tid] = mx;
  __syncthreads();
  if (seg == 0) {
    float m2 = red[tid];
    for (int j = 1; j < 16; ++j) m2 = fmaxf(m2, red[tid + j]);
    rowv[rr] = m2;
  }
  __syncthreads();

  const float rmax = rowv[rr];
  float s = 0.f;
  for (int c = 0; c < 128; ++c) {
    float e = __expf(srow[c] - rmax);
    srow[c] = e;
    s += e;
  }
  red[tid] = s;
  __syncthreads();
  if (seg == 0) {
    float t2 = 0.f;
    for (int j = 0; j < 16; ++j) t2 += red[tid + j];
    rowv[rr] = t2;
  }
  __syncthreads();

  const float inv = 1.0f / rowv[rr];
  float* op = out + (size_t)(b * S + mt * 16 + rr) * 2048 + seg * 128;
  for (int c = 0; c < 128; ++c) op[c] = srow[c] * inv;
}

// ---------------------------------------------------------------------------
// Host launcher.  Workspace layout (f16): hn | q | k | wqT | wkT  (~74.3 MiB)
// ---------------------------------------------------------------------------
extern "C" void kernel_launch(void* const* d_in, const int* in_sizes, int n_in,
                              void* d_out, int out_size, void* d_ws, size_t ws_size,
                              hipStream_t stream) {
  const float* h     = (const float*)d_in[0];
  const float* gamma = (const float*)d_in[1];
  const float* beta  = (const float*)d_in[2];
  const float* wq    = (const float*)d_in[3];
  const float* bq    = (const float*)d_in[4];
  const float* wk    = (const float*)d_in[5];
  const float* bk    = (const float*)d_in[6];
  float* out = (float*)d_out;

  char* ws = (char*)d_ws;
  const size_t HN_BYTES = (size_t)M * D * 2;          // 25,165,824
  const size_t W_BYTES  = (size_t)D * D * 2;          //  1,179,648
  _Float16* hn   = (_Float16*)(ws);
  _Float16* qbuf = (_Float16*)(ws + HN_BYTES);
  _Float16* kbuf = (_Float16*)(ws + 2 * HN_BYTES);
  _Float16* wqT  = (_Float16*)(ws + 3 * HN_BYTES);
  _Float16* wkT  = (_Float16*)(ws + 3 * HN_BYTES + W_BYTES);

  // 0) weight transpose + f16 convert
  const int wgrid = (D * D + 255) / 256;
  wT_f16_kernel<<<wgrid, 256, 0, stream>>>(wq, wqT);
  wT_f16_kernel<<<wgrid, 256, 0, stream>>>(wk, wkT);

  // 1) LayerNorm -> f16
  ln_kernel<<<M, 256, 0, stream>>>(h, gamma, beta, hn);

  // 2) Q and K projections (WMMA): 1024 m-tiles x 12 col-groups = 12288 waves
  gemm_qk_kernel<<<12288 / 8, 256, 0, stream>>>(hn, wqT, bq, qbuf);
  gemm_qk_kernel<<<12288 / 8, 256, 0, stream>>>(hn, wkT, bk, kbuf);

  // 3) fused scores + softmax: one block per (batch, 16-row strip)
  const size_t smem_bytes = 24576 + 131072 + 1024 + 64;   // 156,736 B
  scores_softmax_kernel<<<NB * (S / 16), 256, smem_bytes, stream>>>(qbuf, kbuf, out);
}
